// GraphSenseGAT_79010218377717
// MI455X (gfx1250) — compile-verified, matrix operand order checked by reading, not run on hardware
//
#include <hip/hip_runtime.h>
#include <hip/hip_bf16.h>

typedef __attribute__((ext_vector_type(16))) _Float16 v16h;
typedef __attribute__((ext_vector_type(8)))  float    v8f;

#define N_NODES   65536
#define G_GRAPHS  8192
#define NPG       8
#define IN_F      9
#define HID       64
#define HEADS     4
#define F1        256      // HEADS*HID
#define NEG_SLOPE 0.2f

// f16 A/B fragment of v_wmma_f32_16x16x32_f16 (ISA 7.12.2): per lane the
// 16 halves are two contiguous K-runs of 8: [koff+k0 .. +7] and [koff+16+k0 .. +7]
// with k0 = (lane>=16 ? 8 : 0). With K-contiguous f16 storage this is 2x b128.
__device__ __forceinline__ v16h load_frag(const _Float16* __restrict__ baseK, int k0) {
  union { v16h v; uint4 q[2]; } f;
  f.q[0] = *(const uint4*)(baseK + k0);
  f.q[1] = *(const uint4*)(baseK + k0 + 16);
  return f.v;
}

// CDNA5 async global->LDS copy (16B per lane), tracked by ASYNCcnt (ISA 10.7/15.18.3).
// LDS byte address = low 32 bits of the generic shared pointer (LDS aperture truncation).
__device__ __forceinline__ void async_copy_b128(unsigned lds_byte, unsigned long long gaddr) {
  asm volatile("global_load_async_to_lds_b128 %0, %1, off"
               :: "v"(lds_byte), "v"(gaddr) : "memory");
}
__device__ __forceinline__ void async_wait0() {
  asm volatile("s_wait_asynccnt 0x0" ::: "memory");
}

// ---------------- prep: x (N,9) f32 -> xh (N,32) f16 zero-padded ----------------
__global__ void k_prep_x(const float* __restrict__ x, _Float16* __restrict__ xh) {
  int n = blockIdx.x * blockDim.x + threadIdx.x;
  if (n >= N_NODES) return;
  const float* xr = x + n * IN_F;
  _Float16* o = xh + (size_t)n * 32;
  #pragma unroll
  for (int k = 0; k < 32; ++k) o[k] = (k < IN_F) ? (_Float16)xr[k] : (_Float16)0.f;
}

// ---------------- prep: W (K,N) f32 row-major -> Bt (Npad,Kpad) f16 K-contiguous ----------------
__global__ void k_prep_bt(const float* __restrict__ W, _Float16* __restrict__ Bt,
                          int K, int Kpad, int N, int Npad) {
  int idx = blockIdx.x * blockDim.x + threadIdx.x;   // col*Kpad + k
  if (idx >= Npad * Kpad) return;
  int col = idx / Kpad, k = idx - col * Kpad;
  float v = (k < K && col < N) ? W[k * N + col] : 0.f;
  Bt[idx] = (_Float16)v;
}

// ---------------- GEMM1: xp1 = xh(N,32) @ W1t -> f16 (N,256); 16x64 strip/wave ----------------
__global__ void k_gemm1(const _Float16* __restrict__ xh, const _Float16* __restrict__ W1t,
                        _Float16* __restrict__ xp1h) {
  int lane = threadIdx.x & 31, wid = threadIdx.x >> 5;
  int strip = blockIdx.x * 4 + wid;        // 4096 mtiles x 4 strips
  int mtile = strip >> 2, ns = strip & 3;
  int arow = mtile * 16 + (lane & 15);
  int lcol = lane & 15;
  int k0 = (lane >= 16) ? 8 : 0;
  v16h a = load_frag(xh + (size_t)arow * 32, k0);
  union { v8f v; float f[8]; } c[4];
  #pragma unroll
  for (int nt = 0; nt < 4; ++nt)
    #pragma unroll
    for (int r = 0; r < 8; ++r) c[nt].f[r] = 0.f;
  #pragma unroll
  for (int nt = 0; nt < 4; ++nt) {
    int col = ns * 64 + nt * 16 + lcol;
    v16h b = load_frag(W1t + (size_t)col * 32, k0);
    c[nt].v = __builtin_amdgcn_wmma_f32_16x16x32_f16(false, a, false, b, (short)0, c[nt].v, false, false);
  }
  int rbase = mtile * 16 + ((lane >= 16) ? 8 : 0);
  #pragma unroll
  for (int nt = 0; nt < 4; ++nt) {
    int col = ns * 64 + nt * 16 + lcol;
    #pragma unroll
    for (int r = 0; r < 8; ++r)
      xp1h[(size_t)(rbase + r) * F1 + col] = (_Float16)c[nt].f[r];
  }
}

// ---------------- attention logits precompute, layer 1 ----------------
__global__ void k_al1(const _Float16* __restrict__ xp1h,
                      const float* __restrict__ a_src1, const float* __restrict__ a_dst1,
                      float* __restrict__ als1, float* __restrict__ ald1) {
  int idx = blockIdx.x * blockDim.x + threadIdx.x;   // n*4 + h
  if (idx >= N_NODES * HEADS) return;
  int n = idx >> 2, h = idx & 3;
  const _Float16* xr = xp1h + (size_t)n * F1 + h * HID;
  const float* as = a_src1 + h * HID;
  const float* ad = a_dst1 + h * HID;
  float s = 0.f, d = 0.f;
  #pragma unroll 8
  for (int c = 0; c < HID; ++c) { float v = (float)xr[c]; s += v * as[c]; d += v * ad[c]; }
  als1[idx] = s; ald1[idx] = d;
}

// ---------------- fused attention + aggregate + ELU, layer 1 ----------------
__global__ void k_attn1(const _Float16* __restrict__ xp1h,
                        const float* __restrict__ als1, const float* __restrict__ ald1,
                        const float* __restrict__ ew,
                        const float* __restrict__ We1, const float* __restrict__ ae1,
                        const float* __restrict__ b1,
                        _Float16* __restrict__ h1h) {
  __shared__ __align__(16) _Float16 sxp[8][NPG * F1];
  int lane = threadIdx.x & 31, wid = threadIdx.x >> 5;
  int g = blockIdx.x * 8 + wid;
  int gn0 = g * NPG;
  { // stage graph's xp tile to LDS via async global->LDS (4096B = 8 x 512B waves)
    unsigned lds0 = (unsigned)(size_t)&sxp[wid][0];
    unsigned long long g0 = (unsigned long long)(size_t)(xp1h + (size_t)gn0 * F1);
    #pragma unroll
    for (int t = 0; t < 8; ++t) {
      unsigned idx = (unsigned)(t * 32 + lane) * 16u;
      async_copy_b128(lds0 + idx, g0 + idx);
    }
    async_wait0();
  }
  __syncthreads();
  int dst = lane >> 2, h = lane & 3;
  float ce = 0.f;
  #pragma unroll 8
  for (int c = 0; c < HID; ++c) ce += We1[h * HID + c] * ae1[h * HID + c];
  float wv[NPG]; float wsum = 0.f;
  #pragma unroll
  for (int s = 0; s < NPG; ++s) {
    if (s != dst) {
      int e = g * 56 + s * 7 + (dst - (dst > s ? 1 : 0));
      wv[s] = ew[e]; wsum += wv[s];
    }
  }
  wv[dst] = wsum * (1.0f / 7.0f);
  float ad = ald1[(gn0 + dst) * HEADS + h];
  float lg[NPG]; float mx = -1e30f;
  #pragma unroll
  for (int s = 0; s < NPG; ++s) {
    float l = als1[(gn0 + s) * HEADS + h] + ad + wv[s] * ce;
    l = (l > 0.f) ? l : NEG_SLOPE * l;
    lg[s] = l; mx = fmaxf(mx, l);
  }
  float ssum = 0.f;
  #pragma unroll
  for (int s = 0; s < NPG; ++s) { lg[s] = __expf(lg[s] - mx); ssum += lg[s]; }
  float inv = 1.f / (ssum + 1e-16f);
  for (int c = 0; c < HID; ++c) {
    float o = 0.f;
    #pragma unroll
    for (int s = 0; s < NPG; ++s) o += lg[s] * (float)sxp[wid][s * F1 + h * HID + c];
    o = o * inv + b1[h * HID + c];
    o = (o > 0.f) ? o : (__expf(o) - 1.f);
    h1h[(size_t)(gn0 + dst) * F1 + h * HID + c] = (_Float16)o;
  }
}

// ---------------- GEMM2: xp2 = h1(N,256) @ W2t -> f32 (N,64); 16x64 strip/wave ----------------
__global__ void k_gemm2(const _Float16* __restrict__ h1h, const _Float16* __restrict__ W2t,
                        float* __restrict__ xp2) {
  int lane = threadIdx.x & 31, wid = threadIdx.x >> 5;
  int mtile = blockIdx.x * 4 + wid;        // 4096 mtiles
  int arow = mtile * 16 + (lane & 15);
  int lcol = lane & 15;
  int k0 = (lane >= 16) ? 8 : 0;
  const _Float16* arowp = h1h + (size_t)arow * F1;
  __builtin_prefetch(arowp, 0, 3);         // global_prefetch_b8: pull full A row
  union { v8f v; float f[8]; } c[4];
  #pragma unroll
  for (int nt = 0; nt < 4; ++nt)
    #pragma unroll
    for (int r = 0; r < 8; ++r) c[nt].f[r] = 0.f;
  #pragma unroll
  for (int kk = 0; kk < F1 / 32; ++kk) {
    v16h a = load_frag(arowp + kk * 32, k0);
    #pragma unroll
    for (int nt = 0; nt < 4; ++nt) {
      v16h b = load_frag(W2t + (size_t)(nt * 16 + lcol) * F1 + kk * 32, k0);
      c[nt].v = __builtin_amdgcn_wmma_f32_16x16x32_f16(false, a, false, b, (short)0, c[nt].v, false, false);
    }
  }
  int rbase = mtile * 16 + ((lane >= 16) ? 8 : 0);
  #pragma unroll
  for (int nt = 0; nt < 4; ++nt)
    #pragma unroll
    for (int r = 0; r < 8; ++r)
      xp2[(size_t)(rbase + r) * HID + nt * 16 + lcol] = c[nt].f[r];
}

// ---------------- attention logits precompute, layer 2 ----------------
__global__ void k_al2(const float* __restrict__ xp2,
                      const float* __restrict__ a_src2, const float* __restrict__ a_dst2,
                      float* __restrict__ als2, float* __restrict__ ald2) {
  int n = blockIdx.x * blockDim.x + threadIdx.x;
  if (n >= N_NODES) return;
  const float* xr = xp2 + (size_t)n * HID;
  float s = 0.f, d = 0.f;
  #pragma unroll 8
  for (int c = 0; c < HID; ++c) { float v = xr[c]; s += v * a_src2[c]; d += v * a_dst2[c]; }
  als2[n] = s; ald2[n] = d;
}

// ---------------- fused attention layer 2 + ELU + mean pool ----------------
__global__ void k_attn2(const float* __restrict__ xp2,
                        const float* __restrict__ als2, const float* __restrict__ ald2,
                        const float* __restrict__ ew,
                        const float* __restrict__ We2, const float* __restrict__ ae2,
                        const float* __restrict__ b2,
                        float* __restrict__ emb, _Float16* __restrict__ embh) {
  __shared__ __align__(16) float sxp[8][NPG * HID];
  __shared__ float pool[8][HID];
  int lane = threadIdx.x & 31, wid = threadIdx.x >> 5;
  int g = blockIdx.x * 8 + wid;
  int gn0 = g * NPG;
  { // stage graph's xp2 tile via async global->LDS (2048B = 4 x 512B waves)
    unsigned lds0 = (unsigned)(size_t)&sxp[wid][0];
    unsigned long long g0 = (unsigned long long)(size_t)(xp2 + (size_t)gn0 * HID);
    #pragma unroll
    for (int t = 0; t < 4; ++t) {
      unsigned idx = (unsigned)(t * 32 + lane) * 16u;
      async_copy_b128(lds0 + idx, g0 + idx);
    }
    async_wait0();
  }
  for (int t = lane; t < HID; t += 32) pool[wid][t] = 0.f;
  __syncthreads();
  int dst = lane >> 2, q = lane & 3;
  float ce = 0.f;
  #pragma unroll 8
  for (int c = 0; c < HID; ++c) ce += We2[c] * ae2[c];
  float wv[NPG]; float wsum = 0.f;
  #pragma unroll
  for (int s = 0; s < NPG; ++s) {
    if (s != dst) {
      int e = g * 56 + s * 7 + (dst - (dst > s ? 1 : 0));
      wv[s] = ew[e]; wsum += wv[s];
    }
  }
  wv[dst] = wsum * (1.0f / 7.0f);
  float ad = ald2[gn0 + dst];
  float lg[NPG]; float mx = -1e30f;
  #pragma unroll
  for (int s = 0; s < NPG; ++s) {
    float l = als2[gn0 + s] + ad + wv[s] * ce;
    l = (l > 0.f) ? l : NEG_SLOPE * l;
    lg[s] = l; mx = fmaxf(mx, l);
  }
  float ssum = 0.f;
  #pragma unroll
  for (int s = 0; s < NPG; ++s) { lg[s] = __expf(lg[s] - mx); ssum += lg[s]; }
  float inv = 1.f / (ssum + 1e-16f);
  #pragma unroll
  for (int cc = 0; cc < 16; ++cc) {
    int ch = q * 16 + cc;
    float o = 0.f;
    #pragma unroll
    for (int s = 0; s < NPG; ++s) o += lg[s] * sxp[wid][s * HID + ch];
    o = o * inv + b2[ch];
    o = (o > 0.f) ? o : (__expf(o) - 1.f);
    atomicAdd(&pool[wid][ch], o);
  }
  __syncthreads();
  for (int ch = lane; ch < HID; ch += 32) {
    float v = pool[wid][ch] * 0.125f;
    emb[(size_t)g * HID + ch] = v;
    embh[(size_t)g * HID + ch] = (_Float16)v;
  }
}

// ---------------- GEMM3: t1 = relu(emb(8192,64)@Wr1 + br1) -> f16; 16x64 strip ----------------
__global__ void k_gemm3(const _Float16* __restrict__ embh, const _Float16* __restrict__ Wr1t,
                        const float* __restrict__ br1, _Float16* __restrict__ t1h) {
  int lane = threadIdx.x & 31, wid = threadIdx.x >> 5;
  int strip = blockIdx.x * 4 + wid;        // 512 mtiles x 2 strips
  int mtile = strip >> 1, ns = strip & 1;
  int arow = mtile * 16 + (lane & 15);
  int lcol = lane & 15;
  int k0 = (lane >= 16) ? 8 : 0;
  const _Float16* arowp = embh + (size_t)arow * HID;
  union { v8f v; float f[8]; } c[4];
  #pragma unroll
  for (int nt = 0; nt < 4; ++nt)
    #pragma unroll
    for (int r = 0; r < 8; ++r) c[nt].f[r] = 0.f;
  #pragma unroll
  for (int kk = 0; kk < 2; ++kk) {
    v16h a = load_frag(arowp + kk * 32, k0);
    #pragma unroll
    for (int nt = 0; nt < 4; ++nt) {
      int col = ns * 64 + nt * 16 + lcol;
      v16h b = load_frag(Wr1t + (size_t)col * HID + kk * 32, k0);
      c[nt].v = __builtin_amdgcn_wmma_f32_16x16x32_f16(false, a, false, b, (short)0, c[nt].v, false, false);
    }
  }
  int rbase = mtile * 16 + ((lane >= 16) ? 8 : 0);
  #pragma unroll
  for (int nt = 0; nt < 4; ++nt) {
    int col = ns * 64 + nt * 16 + lcol;
    float bias = br1[col];
    #pragma unroll
    for (int r = 0; r < 8; ++r)
      t1h[(size_t)(rbase + r) * 128 + col] = (_Float16)fmaxf(c[nt].f[r] + bias, 0.f);
  }
}

// ---------------- GEMM4: recon = t1(8192,128)@Wr2 + br2 -> d_out; 16x80 strip ----------------
__global__ void k_gemm4(const _Float16* __restrict__ t1h, const _Float16* __restrict__ Wr2t,
                        const float* __restrict__ br2, float* __restrict__ outr) {
  int lane = threadIdx.x & 31, wid = threadIdx.x >> 5;
  int mtile = blockIdx.x * 4 + wid;        // 512 mtiles, full 80-col strip (72 valid)
  int arow = mtile * 16 + (lane & 15);
  int lcol = lane & 15;
  int k0 = (lane >= 16) ? 8 : 0;
  const _Float16* arowp = t1h + (size_t)arow * 128;
  __builtin_prefetch(arowp, 0, 3);
  union { v8f v; float f[8]; } c[5];
  #pragma unroll
  for (int nt = 0; nt < 5; ++nt)
    #pragma unroll
    for (int r = 0; r < 8; ++r) c[nt].f[r] = 0.f;
  #pragma unroll
  for (int kk = 0; kk < 4; ++kk) {
    v16h a = load_frag(arowp + kk * 32, k0);
    #pragma unroll
    for (int nt = 0; nt < 5; ++nt) {
      v16h b = load_frag(Wr2t + (size_t)(nt * 16 + lcol) * 128 + kk * 32, k0);
      c[nt].v = __builtin_amdgcn_wmma_f32_16x16x32_f16(false, a, false, b, (short)0, c[nt].v, false, false);
    }
  }
  int rbase = mtile * 16 + ((lane >= 16) ? 8 : 0);
  #pragma unroll
  for (int nt = 0; nt < 5; ++nt) {
    int col = nt * 16 + lcol;
    if (col < 72) {
      float bias = br2[col];
      #pragma unroll
      for (int r = 0; r < 8; ++r)
        outr[(size_t)(rbase + r) * 72 + col] = c[nt].f[r] + bias;
    }
  }
}

// ---------------- classifier head ----------------
__global__ void k_cls(const float* __restrict__ emb, const float* __restrict__ Wc1,
                      const float* __restrict__ bc1, const float* __restrict__ Wc2,
                      const float* __restrict__ bc2, float* __restrict__ outc) {
  int g = blockIdx.x * blockDim.x + threadIdx.x;
  if (g >= G_GRAPHS) return;
  const float* e = emb + (size_t)g * HID;
  float acc = bc2[0];
  for (int j = 0; j < 32; ++j) {
    float t = bc1[j];
    #pragma unroll 8
    for (int k = 0; k < HID; ++k) t += e[k] * Wc1[k * 32 + j];
    acc += fmaxf(t, 0.f) * Wc2[j];
  }
  outc[g] = acc;
}

extern "C" void kernel_launch(void* const* d_in, const int* in_sizes, int n_in,
                              void* d_out, int out_size, void* d_ws, size_t ws_size,
                              hipStream_t stream) {
  (void)in_sizes; (void)n_in; (void)out_size; (void)ws_size;
  const float* x      = (const float*)d_in[0];
  const float* ew     = (const float*)d_in[2];
  const float* W1     = (const float*)d_in[4];
  const float* a_src1 = (const float*)d_in[5];
  const float* a_dst1 = (const float*)d_in[6];
  const float* We1    = (const float*)d_in[7];
  const float* a_e1   = (const float*)d_in[8];
  const float* b1     = (const float*)d_in[9];
  const float* W2     = (const float*)d_in[10];
  const float* a_src2 = (const float*)d_in[11];
  const float* a_dst2 = (const float*)d_in[12];
  const float* We2    = (const float*)d_in[13];
  const float* a_e2   = (const float*)d_in[14];
  const float* b2     = (const float*)d_in[15];
  const float* Wr1    = (const float*)d_in[16];
  const float* br1    = (const float*)d_in[17];
  const float* Wr2    = (const float*)d_in[18];
  const float* br2    = (const float*)d_in[19];
  const float* Wc1    = (const float*)d_in[20];
  const float* bc1    = (const float*)d_in[21];
  const float* Wc2    = (const float*)d_in[22];
  const float* bc2    = (const float*)d_in[23];

  char* ws = (char*)d_ws;
  size_t off = 0;
  auto carve = [&](size_t bytes) {
    void* p = ws + off;
    off += (bytes + 255) & ~(size_t)255;
    return p;
  };
  _Float16* xh   = (_Float16*)carve((size_t)N_NODES * 32 * 2);
  _Float16* w1t  = (_Float16*)carve((size_t)F1 * 32 * 2);
  _Float16* w2t  = (_Float16*)carve((size_t)HID * F1 * 2);
  _Float16* wr1t = (_Float16*)carve((size_t)128 * HID * 2);
  _Float16* wr2t = (_Float16*)carve((size_t)80 * 128 * 2);
  _Float16* xp1h = (_Float16*)carve((size_t)N_NODES * F1 * 2);
  _Float16* h1h  = (_Float16*)carve((size_t)N_NODES * F1 * 2);
  float*    als1 = (float*)carve((size_t)N_NODES * HEADS * 4);
  float*    ald1 = (float*)carve((size_t)N_NODES * HEADS * 4);
  float*    xp2  = (float*)carve((size_t)N_NODES * HID * 4);
  float*    als2 = (float*)carve((size_t)N_NODES * 4);
  float*    ald2 = (float*)carve((size_t)N_NODES * 4);
  float*    emb  = (float*)carve((size_t)G_GRAPHS * HID * 4);
  _Float16* embh = (_Float16*)carve((size_t)G_GRAPHS * HID * 2);
  _Float16* t1h  = (_Float16*)carve((size_t)G_GRAPHS * 128 * 2);

  float* outr = (float*)d_out;                 // recon (8192,72)
  float* outc = outr + (size_t)G_GRAPHS * 72;  // cls (8192,)

  // operand packing (f16, K-contiguous B layouts)
  k_prep_x <<<256, 256, 0, stream>>>(x, xh);
  k_prep_bt<<<(F1 * 32 + 255) / 256, 256, 0, stream>>>(W1,  w1t,  IN_F, 32,  F1,  F1);
  k_prep_bt<<<(HID * F1 + 255) / 256, 256, 0, stream>>>(W2,  w2t,  F1,   F1,  HID, HID);
  k_prep_bt<<<(128 * HID + 255) / 256, 256, 0, stream>>>(Wr1, wr1t, HID,  HID, 128, 128);
  k_prep_bt<<<(80 * 128 + 255) / 256, 256, 0, stream>>>(Wr2, wr2t, 128,  128, 72,  80);

  // GAT layer 1
  k_gemm1<<<4096, 128, 0, stream>>>(xh, w1t, xp1h);
  k_al1  <<<1024, 256, 0, stream>>>(xp1h, a_src1, a_dst1, als1, ald1);
  k_attn1<<<1024, 256, 0, stream>>>(xp1h, als1, ald1, ew, We1, a_e1, b1, h1h);
  // GAT layer 2
  k_gemm2<<<1024, 128, 0, stream>>>(h1h, w2t, xp2);
  k_al2  <<<256, 256, 0, stream>>>(xp2, a_src2, a_dst2, als2, ald2);
  k_attn2<<<1024, 256, 0, stream>>>(xp2, als2, ald2, ew, We2, a_e2, b2, emb, embh);
  // heads
  k_gemm3<<<256, 128, 0, stream>>>(embh, wr1t, br1, t1h);
  k_gemm4<<<128, 128, 0, stream>>>(t1h, wr2t, br2, outr);
  k_cls  <<<32, 256, 0, stream>>>(emb, Wc1, bc1, Wc2, bc2, outc);
}
